// PolicyNetwork_39359080301157
// MI455X (gfx1250) — compile-verified, matrix operand order checked by reading, not run on hardware
//
#include <hip/hip_runtime.h>
#include <hip/hip_bf16.h>

#define D_    256
#define TWOD  512
#define NB    128     // batch size B
#define NA    1024    // number of actions N
#define TM    32      // rows per block tile
#define XS    520     // X tile row stride in bf16 elems (16B aligned: 520*2=1040=65*16)
#define H1S   264     // H1 tile row stride in bf16 elems (264*2=528=33*16)

typedef __attribute__((ext_vector_type(16))) __bf16 v16bf;
typedef __attribute__((ext_vector_type(8)))  float  v8f;

union FragBF { v16bf v; uint4 q[2]; };

__device__ __forceinline__ unsigned pk_bf16(float a, float b) {
  unsigned ua = __builtin_bit_cast(unsigned, a);
  unsigned ub = __builtin_bit_cast(unsigned, b);
  ua += 0x7fffu + ((ua >> 16) & 1u);   // round-to-nearest-even
  ub += 0x7fffu + ((ub >> 16) & 1u);
  return (ua >> 16) | (ub & 0xffff0000u);
}
__device__ __forceinline__ unsigned short f2bf(float a) {
  unsigned ua = __builtin_bit_cast(unsigned, a);
  ua += 0x7fffu + ((ua >> 16) & 1u);
  return (unsigned short)(ua >> 16);
}

// ---------------- kernel 1: convert weights f32 -> bf16 ----------------
__global__ __launch_bounds__(256)
void cvt_weights_kernel(const float* __restrict__ w1, const float* __restrict__ w2,
                        unsigned short* __restrict__ w1bf, unsigned short* __restrict__ w2bf) {
  int i = blockIdx.x * 256 + threadIdx.x;   // 0 .. 131071
  w1bf[i] = f2bf(w1[i]);
  w2bf[i] = f2bf(w2[i]);
}

// ---------------- kernel 2: fused gather + 3-layer MLP (bf16 WMMA) ----------------
// grid: 1024 (n) * 4 (row group) blocks, 128 threads (4 waves of 32)
__global__ __launch_bounds__(128)
void mlp_kernel(const float* __restrict__ rf,      // (N, B, D)
                const float* __restrict__ state,   // (1, B, D)
                const int*   __restrict__ actions, // (B, N)
                const float* __restrict__ b1, const float* __restrict__ a1p,
                const float* __restrict__ b2, const float* __restrict__ a2p,
                const float* __restrict__ w3, const float* __restrict__ b3,
                const unsigned short* __restrict__ w1bf,   // (256, 512) bf16
                const unsigned short* __restrict__ w2bf,   // (512, 256) bf16
                float* __restrict__ scores)        // (B, N)
{
  __shared__ unsigned short lds_x [TM * XS];   // 32 x 512 bf16 fusion tile
  __shared__ unsigned short lds_h1[TM * H1S];  // 32 x 256 bf16 hidden tile
  __shared__ float          lds_score[TM];

  const int tid  = threadIdx.x;
  const int lane = tid & 31;
  const int wave = tid >> 5;
  const int n    = blockIdx.x >> 2;   // action index
  const int g    = blockIdx.x & 3;    // batch-row group
  const int row0 = g * TM;            // global batch row base
  const float a1 = a1p[0];
  const float a2 = a2p[0];

  if (tid < TM) lds_score[tid] = 0.0f;

  // ---- build X tile: [state[b,:] , rf[actions[b,n], b, :]] as bf16 ----
  // wave w handles local rows w*8 .. w*8+7
  for (int r = 0; r < 8; ++r) {
    const int lr  = wave * 8 + r;
    const int b   = row0 + lr;
    const int act = actions[b * NA + n];
    const float4* srcS = (const float4*)(state + (size_t)b * D_);
    const float4* srcC = (const float4*)(rf + ((size_t)act * NB + b) * D_);
    #pragma unroll
    for (int it = 0; it < 2; ++it) {
      const int e4 = it * 32 + lane;            // float4 index (0..63)
      float4 v = srcS[e4];
      uint2 p; p.x = pk_bf16(v.x, v.y); p.y = pk_bf16(v.z, v.w);
      *(uint2*)&lds_x[lr * XS + e4 * 4] = p;
      float4 c = srcC[e4];
      uint2 q; q.x = pk_bf16(c.x, c.y); q.y = pk_bf16(c.z, c.w);
      *(uint2*)&lds_x[lr * XS + D_ + e4 * 4] = q;
    }
  }
  __syncthreads();

  // wave -> (row half, column group)
  const int rb    = (wave & 1) * 16;       // row block (16 rows)
  const int cg    = wave >> 1;             // column group
  const int lr16  = lane & 15;
  const int khalf = (lane & 16) ? 8 : 0;   // K sub-offset per ISA bf16 A/B layout
  const int rbase = rb + ((lane & 16) ? 8 : 0); // C fragment row base per lane half

  // ---- layer 1: H1 = prelu(X @ W1^T + b1) ; X 32x512, W1 256x512 ----
  for (int jt = cg * 8; jt < cg * 8 + 8; ++jt) {
    v8f acc = {};
    const int j = jt * 16 + lr16;          // output column / w1 row
    for (int kt = 0; kt < 16; ++kt) {
      const int k0 = kt * 32 + khalf;
      FragBF a, bf;
      a.q[0]  = *(const uint4*)&lds_x[(rb + lr16) * XS + k0];
      a.q[1]  = *(const uint4*)&lds_x[(rb + lr16) * XS + k0 + 16];
      bf.q[0] = *(const uint4*)&w1bf[j * TWOD + k0];
      bf.q[1] = *(const uint4*)&w1bf[j * TWOD + k0 + 16];
      acc = __builtin_amdgcn_wmma_f32_16x16x32_bf16(false, a.v, false, bf.v,
                                                    (short)0, acc, false, false);
    }
    const float bias = b1[j];
    #pragma unroll
    for (int r = 0; r < 8; ++r) {
      float v = acc[r] + bias;
      v = (v >= 0.0f) ? v : a1 * v;
      lds_h1[(rbase + r) * H1S + jt * 16 + lr16] = f2bf(v);
    }
  }
  __syncthreads();

  // ---- layer 2 fused with layer 3: psum = sum_j prelu(H1@W2^T+b2)[.,j]*w3[j] ----
  float psum[8];
  #pragma unroll
  for (int r = 0; r < 8; ++r) psum[r] = 0.0f;

  for (int jt = cg * 16; jt < cg * 16 + 16; ++jt) {
    v8f acc = {};
    const int j = jt * 16 + lr16;          // output column / w2 row
    for (int kt = 0; kt < 8; ++kt) {
      const int k0 = kt * 32 + khalf;
      FragBF a, bf;
      a.q[0]  = *(const uint4*)&lds_h1[(rb + lr16) * H1S + k0];
      a.q[1]  = *(const uint4*)&lds_h1[(rb + lr16) * H1S + k0 + 16];
      bf.q[0] = *(const uint4*)&w2bf[j * D_ + k0];
      bf.q[1] = *(const uint4*)&w2bf[j * D_ + k0 + 16];
      acc = __builtin_amdgcn_wmma_f32_16x16x32_bf16(false, a.v, false, bf.v,
                                                    (short)0, acc, false, false);
    }
    const float bias = b2[j];
    const float wj   = w3[j];
    #pragma unroll
    for (int r = 0; r < 8; ++r) {
      float v = acc[r] + bias;
      v = (v >= 0.0f) ? v : a2 * v;
      psum[r] += v * wj;
    }
  }

  // reduce psum across each 16-lane half (xor masks stay inside the half)
  #pragma unroll
  for (int r = 0; r < 8; ++r) {
    float s = psum[r];
    s += __shfl_xor(s, 1, 32);
    s += __shfl_xor(s, 2, 32);
    s += __shfl_xor(s, 4, 32);
    s += __shfl_xor(s, 8, 32);
    if ((lane & 15) == 0) {
      atomicAdd(&lds_score[rbase + r], s);   // two column-group waves combine here
    }
  }
  __syncthreads();

  if (tid < TM) {
    const int b = row0 + tid;
    scores[(size_t)b * NA + n] = lds_score[tid] + b3[0];
  }
}

// ---------------- kernel 3: per-batch softmax + argmax + scatter ----------------
__global__ __launch_bounds__(256)
void softmax_kernel(const float* __restrict__ scores,  // (B, N)
                    const int*   __restrict__ actions, // (B, N)
                    const int*   __restrict__ step_p,
                    float* __restrict__ out)           // [B result | B score | B*N new_actions]
{
  __shared__ float smax[256];
  __shared__ int   sidx[256];
  __shared__ float ssum[256];
  const int b = blockIdx.x;
  const int t = threadIdx.x;
  const float* srow = scores + (size_t)b * NA;
  const int*   arow = actions + (size_t)b * NA;

  float best = -3.4e38f; int bi = NA;
  for (int i = t; i < NA; i += 256) {
    float v = srow[i];
    if (v > best) { best = v; bi = i; }
  }
  smax[t] = best; sidx[t] = bi;
  __syncthreads();
  for (int off = 128; off > 0; off >>= 1) {
    if (t < off) {
      float v2 = smax[t + off]; int i2 = sidx[t + off];
      if (v2 > smax[t] || (v2 == smax[t] && i2 < sidx[t])) { smax[t] = v2; sidx[t] = i2; }
    }
    __syncthreads();
  }
  const float mx    = smax[0];
  const int  sample = sidx[0];

  float ps = 0.0f;
  for (int i = t; i < NA; i += 256) ps += __expf(srow[i] - mx);
  ssum[t] = ps;
  __syncthreads();
  for (int off = 128; off > 0; off >>= 1) {
    if (t < off) ssum[t] += ssum[t + off];
    __syncthreads();
  }

  float* na = out + 2 * NB;
  for (int i = t; i < NA; i += 256) na[(size_t)b * NA + i] = (float)arow[i];
  __syncthreads();
  if (t == 0) {
    const int cr = arow[sample];               // chosen action value
    out[b]      = (float)cr;                   // choosed_result
    out[NB + b] = 1.0f / ssum[0];              // probs[b, sample] = exp(0)/sumexp
    na[(size_t)b * NA + cr] = -(float)step_p[0]; // new_actions scatter at column = value
  }
}

extern "C" void kernel_launch(void* const* d_in, const int* in_sizes, int n_in,
                              void* d_out, int out_size, void* d_ws, size_t ws_size,
                              hipStream_t stream) {
  const float* rf      = (const float*)d_in[0];   // result_features (N,B,D)
  const float* state   = (const float*)d_in[1];   // (1,B,D)
  const int*   actions = (const int*)d_in[2];     // (B,N)
  const int*   step    = (const int*)d_in[3];
  const float* w1      = (const float*)d_in[4];   // (256,512)
  const float* b1      = (const float*)d_in[5];
  const float* a1      = (const float*)d_in[6];
  const float* w2      = (const float*)d_in[7];   // (512,256)
  const float* b2      = (const float*)d_in[8];
  const float* a2      = (const float*)d_in[9];
  const float* w3      = (const float*)d_in[10];  // (1,512)
  const float* b3      = (const float*)d_in[11];

  // workspace layout: w1bf (256KB) | w2bf (256KB) | scores (512KB)
  unsigned short* w1bf = (unsigned short*)d_ws;
  unsigned short* w2bf = w1bf + D_ * TWOD;
  float* scores = (float*)((char*)d_ws + 2 * (size_t)(D_ * TWOD) * sizeof(unsigned short));

  cvt_weights_kernel<<<(D_ * TWOD) / 256, 256, 0, stream>>>(w1, w2, w1bf, w2bf);
  mlp_kernel<<<NA * 4, 128, 0, stream>>>(rf, state, actions, b1, a1, b2, a2,
                                         w3, b3, w1bf, w2bf, scores);
  softmax_kernel<<<NB, 256, 0, stream>>>(scores, actions, step, (float*)d_out);
}
